// BackwardCompatibleLoss_50345606644301
// MI455X (gfx1250) — matrix operand order repaired and checked by reading, not compiled, not run on hardware
//
#include <hip/hip_runtime.h>
#include <hip/hip_bf16.h>

// CDNA5 (gfx1250) wave32 WMMA types
typedef __attribute__((ext_vector_type(16))) _Float16 v16h;
typedef __attribute__((ext_vector_type(8)))  float    v8f;

#define BROWS 4096
#define DDIM  512
#define TEMP_INV 100.0f          // 1 / 0.01
#define NEGBIG  (-3.0e38f)

union FragA { v16h v; uint4 u[2]; };

// ---------------------------------------------------------------------------
// Kernel 1: row-wise L2 normalize feat & feat_old, emit f16 copies + positive
// logit  pos[i] = <fn_i, fo_i>  computed fully in f32.
// ---------------------------------------------------------------------------
__global__ __launch_bounds__(256) void bcl_norm_kernel(
    const float* __restrict__ feat, const float* __restrict__ feat_old,
    _Float16* __restrict__ fnh, _Float16* __restrict__ foh,
    float* __restrict__ pos) {
  const int row = blockIdx.x;
  const int t   = threadIdx.x;
  const float* a = feat     + (size_t)row * DDIM;
  const float* b = feat_old + (size_t)row * DDIM;
  float x0 = a[t], x1 = a[t + 256];
  float y0 = b[t], y1 = b[t + 256];
  float saa = x0 * x0 + x1 * x1;
  float sbb = y0 * y0 + y1 * y1;
  float sab = x0 * y0 + x1 * y1;
  #pragma unroll
  for (int off = 16; off; off >>= 1) {
    saa += __shfl_xor(saa, off, 32);
    sbb += __shfl_xor(sbb, off, 32);
    sab += __shfl_xor(sab, off, 32);
  }
  __shared__ float r0[8], r1[8], r2[8];
  const int wave = t >> 5, lane = t & 31;
  if (lane == 0) { r0[wave] = saa; r1[wave] = sbb; r2[wave] = sab; }
  __syncthreads();
  saa = 0.f; sbb = 0.f; sab = 0.f;
  #pragma unroll
  for (int w = 0; w < 8; ++w) { saa += r0[w]; sbb += r1[w]; sab += r2[w]; }
  const float sa = 1.0f / fmaxf(sqrtf(saa), 1e-12f);
  const float sb = 1.0f / fmaxf(sqrtf(sbb), 1e-12f);
  if (t == 0) pos[row] = sab * sa * sb;
  _Float16* fa = fnh + (size_t)row * DDIM;
  _Float16* fb = foh + (size_t)row * DDIM;
  fa[t]       = (_Float16)(x0 * sa);
  fa[t + 256] = (_Float16)(x1 * sa);
  fb[t]       = (_Float16)(y0 * sb);
  fb[t + 256] = (_Float16)(y1 * sb);
}

// ---------------------------------------------------------------------------
// Kernel 2: per 16-row tile, stream all 256 column tiles of BOTH similarity
// matrices through v_wmma_f32_16x16x32_f16 with an online (max, sum-exp),
// masking same-label pairs.
//
// Depth-4 register pipeline on A fragments, pinned with sched_barrier(0) so
// the scheduler cannot re-sink the prefetches; the last 4 prefetch slots of
// each tile fetch chunks 0..3 of the NEXT tile so its preamble overlaps the
// current tile's softmax epilogue.
// ---------------------------------------------------------------------------
__global__ __launch_bounds__(256) void bcl_loss_kernel(
    const _Float16* __restrict__ fnh, const _Float16* __restrict__ foh,
    const int* __restrict__ targets, const float* __restrict__ pos,
    float* __restrict__ partial) {
  __shared__ int    tLds[BROWS];
  __shared__ float4 red[8][16];

  const int tid = threadIdx.x;
  for (int i = tid; i < BROWS; i += 256) tLds[i] = targets[i];
  __syncthreads();

  const int wave = tid >> 5;
  const int lane = tid & 31;
  const int ln   = lane & 15;
  const int hi   = lane >> 4;
  const int r0   = blockIdx.x * 16;
  const int myRow = r0 + ln;
  const int tRow  = tLds[myRow];

  // resident B fragments: the 16 fn target rows, all 16 K-chunks (128 VGPRs)
  v16h bF[16];
  {
    const _Float16* bbase = fnh + (size_t)myRow * DDIM + 8 * hi;
    #pragma unroll
    for (int c = 0; c < 16; ++c) {
      FragA f;
      f.u[0] = *(const uint4*)(bbase + 32 * c);
      f.u[1] = *(const uint4*)(bbase + 32 * c + 16);
      bF[c] = f.v;
    }
  }

  float m1 = NEGBIG, s1 = 0.f;   // new-to-old
  float m2 = NEGBIG, s2 = 0.f;   // new-to-new

  const size_t tileStep = (size_t)128 * DDIM;   // 8 waves x 16 rows
  const _Float16* aO0 = foh + (size_t)(wave * 16 + ln) * DDIM + 8 * hi;
  const _Float16* aN0 = fnh + (size_t)(wave * 16 + ln) * DDIM + 8 * hi;
  const _Float16* aO = aO0;
  const _Float16* aN = aN0;

  // preamble: chunks 0..3 of the first tile
  FragA pO[4], pN[4];
  #pragma unroll
  for (int c = 0; c < 4; ++c) {
    pO[c].u[0] = *(const uint4*)(aO + 32 * c);
    pO[c].u[1] = *(const uint4*)(aO + 32 * c + 16);
    pN[c].u[0] = *(const uint4*)(aN + 32 * c);
    pN[c].u[1] = *(const uint4*)(aN + 32 * c + 16);
  }

  for (int t = 0; t < BROWS / 16 / 8; ++t) {     // 32 tiles per wave
    const int j0 = (wave + 8 * t) * 16;
    const _Float16* aOn = (t == 31) ? aO0 : (aO + tileStep);  // wrap: dummy
    const _Float16* aNn = (t == 31) ? aN0 : (aN + tileStep);

    v8f accO = {}, accN = {};
    #pragma unroll
    for (int c = 0; c < 16; ++c) {
      accO = __builtin_amdgcn_wmma_f32_16x16x32_f16(
          false, pO[c & 3].v, false, bF[c], (short)0, accO, false, false);
      accN = __builtin_amdgcn_wmma_f32_16x16x32_f16(
          false, pN[c & 3].v, false, bF[c], (short)0, accN, false, false);
      // prefetch distance 4; last 4 slots fetch next tile's chunks 0..3
      const _Float16* srcO = (c < 12) ? (aO + 32 * (c + 4)) : (aOn + 32 * (c - 12));
      const _Float16* srcN = (c < 12) ? (aN + 32 * (c + 4)) : (aNn + 32 * (c - 12));
      pO[c & 3].u[0] = *(const uint4*)(srcO);
      pO[c & 3].u[1] = *(const uint4*)(srcO + 16);
      pN[c & 3].u[0] = *(const uint4*)(srcN);
      pN[c & 3].u[1] = *(const uint4*)(srcN + 16);
      __builtin_amdgcn_sched_barrier(0);   // pin: loads may not sink
    }

    // per-tile labels: 8 contiguous ints -> two ds_load_b128
    const int4 ta = *(const int4*)&tLds[j0 + 8 * hi];
    const int4 tb = *(const int4*)&tLds[j0 + 8 * hi + 4];
    const int tn[8] = {ta.x, ta.y, ta.z, ta.w, tb.x, tb.y, tb.z, tb.w};

    // masked online softmax, single exp per element:
    //   e = exp(min(m,v) - max(m,v));  s = v>m ? s*e+1 : s+e
    #pragma unroll
    for (int v = 0; v < 8; ++v) {
      const bool eq = (tn[v] == tRow);
      {
        const float vv = eq ? NEGBIG : accO[v] * TEMP_INV;
        const float mn = fmaxf(m1, vv);
        const float e  = __expf(fminf(m1, vv) - mn);
        s1 = (vv > m1) ? (s1 * e + 1.0f) : (s1 + (eq ? 0.f : e));
        m1 = mn;
      }
      {
        const float vv = eq ? NEGBIG : accN[v] * TEMP_INV;
        const float mn = fmaxf(m2, vv);
        const float e  = __expf(fminf(m2, vv) - mn);
        s2 = (vv > m2) ? (s2 * e + 1.0f) : (s2 + (eq ? 0.f : e));
        m2 = mn;
      }
    }
    aO = aOn;
    aN = aNn;
  }

  // fold lane L with lane L^16 (both hold partials for the same row)
  {
    float mo = __shfl_xor(m1, 16, 32), so = __shfl_xor(s1, 16, 32);
    float mn = fmaxf(m1, mo);
    s1 = s1 * __expf(m1 - mn) + so * __expf(mo - mn); m1 = mn;
    mo = __shfl_xor(m2, 16, 32); so = __shfl_xor(s2, 16, 32);
    mn = fmaxf(m2, mo);
    s2 = s2 * __expf(m2 - mn) + so * __expf(mo - mn); m2 = mn;
  }
  if (hi == 0) red[wave][ln] = make_float4(m1, s1, m2, s2);
  __syncthreads();

  if (tid < 16) {
    float M = NEGBIG, S = 0.f;
    #pragma unroll
    for (int w = 0; w < 8; ++w) {
      const float4 q = red[w][tid];
      float mn = fmaxf(M, q.x);
      S = S * __expf(M - mn) + q.y * __expf(q.x - mn); M = mn;
      mn = fmaxf(M, q.z);
      S = S * __expf(M - mn) + q.w * __expf(q.z - mn); M = mn;
    }
    const float pv = pos[r0 + tid] * TEMP_INV;
    const float mn = fmaxf(M, pv);
    const float tot = S * __expf(M - mn) + __expf(pv - mn);
    float rowLoss = mn + __logf(tot) - pv;   // lse - logit[0]
    #pragma unroll
    for (int off = 8; off; off >>= 1) rowLoss += __shfl_xor(rowLoss, off, 32);
    if (tid == 0) partial[blockIdx.x] = rowLoss;
  }
}

// ---------------------------------------------------------------------------
// Kernel 3: deterministic reduction of the 256 per-block partials -> mean.
// ---------------------------------------------------------------------------
__global__ __launch_bounds__(256) void bcl_final_kernel(
    const float* __restrict__ partial, float* __restrict__ out) {
  const int t = threadIdx.x;
  float v = partial[t];
  #pragma unroll
  for (int off = 16; off; off >>= 1) v += __shfl_xor(v, off, 32);
  __shared__ float r[8];
  if ((t & 31) == 0) r[t >> 5] = v;
  __syncthreads();
  if (t == 0) {
    float s = 0.f;
    #pragma unroll
    for (int w = 0; w < 8; ++w) s += r[w];
    out[0] = s * (1.0f / BROWS);   // mean, LOSS_WEIGHT == 1
  }
}

extern "C" void kernel_launch(void* const* d_in, const int* in_sizes, int n_in,
                              void* d_out, int out_size, void* d_ws, size_t ws_size,
                              hipStream_t stream) {
  (void)in_sizes; (void)n_in; (void)out_size; (void)ws_size;
  const float* feat     = (const float*)d_in[0];
  const float* feat_old = (const float*)d_in[1];
  const int*   targets  = (const int*)d_in[2];
  float* out = (float*)d_out;

  char* ws = (char*)d_ws;
  _Float16* fnh = (_Float16*)ws;                                    // 4 MB
  _Float16* foh = (_Float16*)(ws + (size_t)BROWS * DDIM * 2);       // 4 MB
  float* pos     = (float*)(ws + (size_t)BROWS * DDIM * 4);         // 16 KB
  float* partial = pos + BROWS;                                     // 1 KB

  bcl_norm_kernel<<<BROWS, 256, 0, stream>>>(feat, feat_old, fnh, foh, pos);
  bcl_loss_kernel<<<BROWS / 16, 256, 0, stream>>>(fnh, foh, targets, pos, partial);
  bcl_final_kernel<<<1, 256, 0, stream>>>(partial, out);
}
